// SCNN_86062554677964
// MI455X (gfx1250) — compile-verified, matrix operand order checked by reading, not run on hardware
//
#include <hip/hip_runtime.h>

// ---------------------------------------------------------------------------
// MI455X (gfx1250) message-passing scan.
//
// Dependency-bound: 540 sequential conv-GEMM steps; tensor (118 MB) is
// L2-resident (192 MB L2).  One persistent workgroup per batch (16 WGs),
// recurrent row state double-buffered in LDS as bf16 (2 x 57 KB of the
// 320 KB WGP LDS), each step executed as v_wmma_f32_16x16x32_bf16 GEMMs
// with pre-packed A fragments.  512 threads / 16 waves per WG keeps VGPR
// pressure low enough that no operand ever touches scratch.
// ---------------------------------------------------------------------------

typedef __bf16 v16bf __attribute__((ext_vector_type(16)));
typedef float  v8f   __attribute__((ext_vector_type(8)));
typedef int    v8i   __attribute__((ext_vector_type(8)));

#define C_CH 128
#define HH   72
#define WW   200
#define HWSZ (HH * WW)
#define CHW  (C_CH * HWSZ)
#define KTAP 9
#define NFRAG_PER_PASS (8 * KTAP * 4)   // co-tiles(8) * taps(9) * ci-chunks(4) = 288
#define FRAG_ELEMS 512                  // 16x32 bf16 fragment
#define ROWPITCH_DW 66                  // 132 bf16 per LDS row (128 + pad, bank spread)
#define MAXROWS 216                     // 13 tiles * 16 + 8 halo rows (vertical case)
#define NTHREADS 512                    // 16 waves

// ---------------------------------------------------------------------------
// Pre-pack the four (128,128,9) fp32 weight tensors into WMMA A-fragment
// layout, bf16.  Fragment (ct, kp, kkc) is the 16x32 slice
// W[ct*16 : ct*16+16, kkc*32 : kkc*32+32] of the per-tap matrix W[:, :, kp].
// Per-lane element order matches the 16-bit A-matrix 16x32 VGPR layout:
//   lane L (g = L>>4, m = L&15), slot s (0..15):  k = s + 8g + (s>=8 ? 8 : 0)
// so each lane does one contiguous 32-byte load at runtime.
// ---------------------------------------------------------------------------
__global__ void pack_weights_kernel(const float* __restrict__ w0,
                                    const float* __restrict__ w1,
                                    const float* __restrict__ w2,
                                    const float* __restrict__ w3,
                                    __bf16* __restrict__ packed) {
  int gid = blockIdx.x * blockDim.x + threadIdx.x;
  const int total = 4 * NFRAG_PER_PASS * FRAG_ELEMS;
  if (gid >= total) return;
  int e = gid & (FRAG_ELEMS - 1);
  int f = (gid >> 9) % NFRAG_PER_PASS;
  int p = (gid >> 9) / NFRAG_PER_PASS;
  int lane = e >> 4, s = e & 15;
  int kkc = f & 3;
  int kp  = (f >> 2) % KTAP;
  int ct  = (f >> 2) / KTAP;
  int m = lane & 15, g = lane >> 4;
  int k = s + 8 * g + (s >= 8 ? 8 : 0);
  int co = ct * 16 + m;
  int ci = kkc * 32 + k;
  const float* w = (p == 0) ? w0 : (p == 1) ? w1 : (p == 2) ? w2 : w3;
  packed[gid] = (__bf16)w[(co * C_CH + ci) * KTAP + kp];
}

// ---------------------------------------------------------------------------
// One scan sweep.  grid = 16 (one WG per batch), block = 512 (16 waves).
//
//   idx(b,c,t,pos) = b*CHW + c*HWSZ + t*stepStride + pos*posStride
//   vertical:   T=H, L=W, stepStride=W, posStride=1
//   horizontal: T=W, L=H, stepStride=1, posStride=W
//
// Double-buffered LDS recurrent state prev[buf][pos][ci] (bf16, 4-row zero
// halo each side, padded pitch for bank spread).  Step t reads buf (t+1)&1,
// writes buf t&1 -> single barrier per step, no cross-tile accumulator
// arrays, no spills.  Wave w owns channel-tile ct = w&7 and strides over
// column tiles by 2, so its 36 A fragments stay L0-hot across tiles.
// ---------------------------------------------------------------------------
__global__ __launch_bounds__(NTHREADS, 1)
void sweep_kernel(const float* in, float* out,
                  const __bf16* __restrict__ wpack,
                  int T, int L, int stepStride, int posStride, int reverse) {
  __shared__ int prevd[2][MAXROWS * ROWPITCH_DW];

  const int tid  = threadIdx.x;
  const int wave = tid >> 5;
  const int lane = tid & 31;
  const int n    = lane & 15;
  const int g    = lane >> 4;
  const int ct   = wave & 7;       // owned channel tile
  const int wt0  = wave >> 3;      // column-tile start (0 or 1), stride 2
  const int b    = blockIdx.x;
  const long bbase = (long)b * CHW;
  const int LT = (L + 15) >> 4;    // col tiles: 13 (vertical) / 5 (horizontal)

  // Zero both LDS buffers (halo + overhang rows stay zero forever).
  for (int i = tid; i < 2 * MAXROWS * ROWPITCH_DW; i += NTHREADS)
    prevd[0][i] = 0;
  __syncthreads();

  // t = 0: first slice passes through unchanged; seed LDS buffer 0.
  {
    int t0 = reverse ? (T - 1) : 0;
    long sbase = bbase + (long)t0 * stepStride;
    for (int i = tid; i < 64 * L; i += NTHREADS) {
      int c2 = i / L, pos = i - c2 * L;
      long o = sbase + (long)pos * posStride + (long)(2 * c2) * HWSZ;
      float v0 = in[o];
      float v1 = in[o + HWSZ];
      out[o] = v0;
      out[o + HWSZ] = v1;
      unsigned short h0 = __builtin_bit_cast(unsigned short, (__bf16)v0);
      unsigned short h1 = __builtin_bit_cast(unsigned short, (__bf16)v1);
      prevd[0][(pos + 4) * ROWPITCH_DW + c2] = (int)h0 | ((int)h1 << 16);
    }
  }

  for (int t = 1; t < T; ++t) {
    __syncthreads();  // step t-1 state fully written
    const int rbuf = (t + 1) & 1;   // holds y_{t-1}
    const int wbuf = t & 1;         // receives y_t
    int tphys = reverse ? (T - 1 - t) : t;
    long sbase = bbase + (long)tphys * stepStride;

    #pragma unroll 1
    for (int wt = wt0; wt < LT; wt += 2) {
      int wbase = wt << 4;
      v8f acc = {};
      #pragma unroll 1
      for (int kp = 0; kp < KTAP; ++kp) {
        // LDS row for B element (pos + kp - 4) with +4 halo offset
        int rowoff = (wbase + n + kp) * ROWPITCH_DW;
        #pragma unroll
        for (int kkc = 0; kkc < 4; ++kkc) {
          // A fragment: contiguous 32B per lane from packed weights (L0/L2-hot)
          v16bf a = *((const v16bf*)wpack +
                      (long)((((ct * KTAP + kp) << 2) + kkc) * 32 + lane));
          // B fragment: 8 dwords, each a packed pair of consecutive-ci bf16
          v8i bi;
          #pragma unroll
          for (int d = 0; d < 8; ++d) {
            int k0 = 2 * d + 8 * g + ((d >= 4) ? 8 : 0);  // even k of slot pair
            bi[d] = prevd[rbuf][rowoff + (kkc << 4) + (k0 >> 1)];
          }
          v16bf bb = __builtin_bit_cast(v16bf, bi);
          acc = __builtin_amdgcn_wmma_f32_16x16x32_bf16(
              false, a, false, bb, (short)0, acc, false, false);
        }
      }

      // Epilogue: y = cur + relu(acc); write global + new LDS state.
      int pos = wbase + n;
      if (pos < L) {
        long o = sbase + (long)pos * posStride;
        #pragma unroll
        for (int r = 0; r < 8; r += 2) {
          int c0 = ct * 16 + 8 * g + r;  // C/D layout: vgpr r -> M = r + 8g
          long oc = o + (long)c0 * HWSZ;
          float y0 = in[oc]        + fmaxf(acc[r],     0.0f);
          float y1 = in[oc + HWSZ] + fmaxf(acc[r + 1], 0.0f);
          out[oc] = y0;
          out[oc + HWSZ] = y1;
          unsigned short h0 = __builtin_bit_cast(unsigned short, (__bf16)y0);
          unsigned short h1 = __builtin_bit_cast(unsigned short, (__bf16)y1);
          prevd[wbuf][(pos + 4) * ROWPITCH_DW + (c0 >> 1)] =
              (int)h0 | ((int)h1 << 16);
        }
      }
    }

    // Prefetch next step's `cur` slab into cache (global_prefetch_b8).
    if (t + 1 < T) {
      int tnext = reverse ? (T - 2 - t) : (t + 1);
      long nb = bbase + (long)tnext * stepStride;
      int total = C_CH * L;
      int chunk = (total + NTHREADS - 1) / NTHREADS;
      int j = tid * chunk;
      if (j < total) {
        int c = j / L, pos = j - c * L;
        __builtin_prefetch(in + nb + (long)c * HWSZ + (long)pos * posStride, 0, 1);
      }
    }
  }
}

// ---------------------------------------------------------------------------
extern "C" void kernel_launch(void* const* d_in, const int* in_sizes, int n_in,
                              void* d_out, int out_size, void* d_ws, size_t ws_size,
                              hipStream_t stream) {
  const float* x   = (const float*)d_in[0];
  const float* wv1 = (const float*)d_in[1];
  const float* wv2 = (const float*)d_in[2];
  const float* wh1 = (const float*)d_in[3];
  const float* wh2 = (const float*)d_in[4];
  float* out = (float*)d_out;
  __bf16* wpack = (__bf16*)d_ws;  // 4 * 288 KiB of packed bf16 fragments

  const int total = 4 * NFRAG_PER_PASS * FRAG_ELEMS;
  pack_weights_kernel<<<(total + 255) / 256, 256, 0, stream>>>(wv1, wv2, wh1, wh2, wpack);

  const int PASS = NFRAG_PER_PASS * FRAG_ELEMS;
  // vertical forward / backward: scan over H, rows of length W
  sweep_kernel<<<16, NTHREADS, 0, stream>>>(x,   out, wpack + 0L * PASS, HH, WW, WW, 1, 0);
  sweep_kernel<<<16, NTHREADS, 0, stream>>>(out, out, wpack + 1L * PASS, HH, WW, WW, 1, 1);
  // horizontal forward / backward: scan over W, rows of length H
  sweep_kernel<<<16, NTHREADS, 0, stream>>>(out, out, wpack + 2L * PASS, WW, HH, 1, WW, 0);
  sweep_kernel<<<16, NTHREADS, 0, stream>>>(out, out, wpack + 3L * PASS, WW, HH, 1, WW, 1);
}
